// ConceptBank_19387482374327
// MI455X (gfx1250) — matrix operand order repaired and checked by reading, not compiled, not run on hardware
//
#include <hip/hip_runtime.h>
#include <hip/hip_bf16.h>

typedef __attribute__((ext_vector_type(16))) __bf16 v16bf;
typedef __attribute__((ext_vector_type(8)))  float  v8f;

#define NKEYS   500
#define DIM     768
#define KD      128
#define MTILE   16
#define NTHREADS 512

// ---- WMMA operand loaders (layouts per CDNA5 ISA 7.12.2) ----

// A (16x32 bf16) from LDS, row-major with given stride.
// lane: m = lane&15, half = lane>>4; element e -> K = e + (e&8) + 8*half
__device__ __forceinline__ v16bf ldsA(const __bf16* base, int stride, int m, int k0, int half) {
  v16bf a;
#pragma unroll
  for (int e = 0; e < 16; ++e) {
    int kk = e + (e & 8) + 8 * half;
    a[e] = base[m * stride + k0 + kk];
  }
  return a;
}

// B (32x16 bf16) from global fp32 row-major [K][N]: lane = N col, element e -> K = e + (e&8) + 8*half.
__device__ __forceinline__ v16bf ldB_rowmajor(const float* W, int ldw, int k0, int n, int half) {
  v16bf b;
  const float* p = W + n;
#pragma unroll
  for (int e = 0; e < 16; ++e) b[e] = (__bf16)p[(size_t)(k0 + e + (e & 8) + 8 * half) * ldw];
  return b;
}

// B for keys^T: B[k][n] = keys[n][k]; keys is [NKEYS][64] row-major. Guard n>=NKEYS -> 0.
__device__ __forceinline__ v16bf ldB_keysT(const float* keys, int k0, int n, int half) {
  v16bf b;
#pragma unroll
  for (int e = 0; e < 16; ++e) {
    int k = k0 + e + (e & 8) + 8 * half;
    int nc = (n < NKEYS) ? n : (NKEYS - 1);
    float v = keys[nc * 64 + k];
    b[e] = (__bf16)((n < NKEYS) ? v : 0.0f);
  }
  return b;
}

__device__ __forceinline__ void ins4(float* bv, int* bi, float sv, int idx) {
  if (sv > bv[3]) {
    bv[3] = sv; bi[3] = idx;
#pragma unroll
    for (int j = 3; j > 0; --j) {
      if (bv[j] > bv[j - 1]) {
        float tv = bv[j]; bv[j] = bv[j - 1]; bv[j - 1] = tv;
        int   ti = bi[j]; bi[j] = bi[j - 1]; bi[j - 1] = ti;
      } else break;
    }
  }
}

__device__ __forceinline__ void top4_scan(const float* row, float* bv, int* bi) {
  bv[0] = bv[1] = bv[2] = bv[3] = -3.4e38f;
  bi[0] = bi[1] = bi[2] = bi[3] = 0;
  for (int n = 0; n < NKEYS; ++n) ins4(bv, bi, row[n], n);
}

__global__ __launch_bounds__(NTHREADS, 1)
void concept_bank_fused(const float* __restrict__ regs,
                        const float* __restrict__ row_keys,
                        const float* __restrict__ col_keys,
                        const float* __restrict__ concepts,
                        const float* __restrict__ gamma,
                        const float* __restrict__ beta,
                        const float* __restrict__ Wq,
                        const float* __restrict__ bq,
                        const float* __restrict__ Wqc,
                        const float* __restrict__ bqc,
                        float* __restrict__ out0,      // (B, R*4, 768) concepts
                        float* __restrict__ out_idx,   // (B, R, 4) as float
                        float* __restrict__ out_sc,    // (B, R, 4)
                        float* __restrict__ out3) {    // (B, R, 768) projection
  // 32 KB region: first used as bf16 LN output [16][768], later as f32 scores [16][512]
  __shared__ __align__(16) unsigned char s_buf[MTILE * 512 * 4];
  __shared__ __align__(16) __bf16 s_q[MTILE * KD];   // queries, bf16
  __shared__ int s_tki[MTILE * 4];

  __bf16* xbf = (__bf16*)s_buf;
  float*  sc  = (float*)s_buf;

  const int tid  = threadIdx.x;
  const int lane = tid & 31;
  const int wv   = tid >> 5;           // 16 waves
  const int half = lane >> 4;
  const int ml   = lane & 15;
  const int nl   = lane & 15;
  const int m0   = blockIdx.x * MTILE; // global query-row base

  // ---------------- Phase LN: wave wv normalizes row wv ----------------
  {
    const float* x = regs + (size_t)(m0 + wv) * DIM;
    float vals[24];
    float s = 0.0f, ss = 0.0f;
#pragma unroll
    for (int i = 0; i < 24; ++i) {
      float v = x[lane + 32 * i];
      vals[i] = v; s += v; ss += v * v;
    }
#pragma unroll
    for (int off = 16; off >= 1; off >>= 1) {
      s  += __shfl_xor(s,  off, 32);
      ss += __shfl_xor(ss, off, 32);
    }
    const float inv = 1.0f / (float)DIM;
    float mu  = s * inv;
    float var = ss * inv - mu * mu;
    float rs  = rsqrtf(var + 1e-5f);
#pragma unroll
    for (int i = 0; i < 24; ++i) {
      int c = lane + 32 * i;
      xbf[wv * DIM + c] = (__bf16)((vals[i] - mu) * rs * gamma[c] + beta[c]);
    }
  }
  __syncthreads();

  // ---------------- Phase Q: queries = LN @ Wq + bq (M=16, N=128, K=768) ----------------
  if (wv < 8) {  // wave-uniform branch; EXEC all-ones inside
    const int n0 = wv * 16;
    v8f acc = {0.f,0.f,0.f,0.f,0.f,0.f,0.f,0.f};
    for (int k0 = 0; k0 < DIM; k0 += 32) {
      v16bf a = ldsA(xbf, DIM, ml, k0, half);
      v16bf b = ldB_rowmajor(Wq, KD, k0, n0 + nl, half);
      acc = __builtin_amdgcn_wmma_f32_16x16x32_bf16(false, a, false, b, (short)0, acc, false, false);
    }
#pragma unroll
    for (int r = 0; r < 8; ++r) {
      int m = r + 8 * half;
      s_q[m * KD + n0 + nl] = (__bf16)(acc[r] + bq[n0 + nl]);
    }
  }
  __syncthreads();

  // ---------------- Score pass helper: q[:, qoff:qoff+64] @ keys^T -> sc[16][512] ----------------
  auto score_pass = [&](const float* keys, int qoff) {
#pragma unroll
    for (int i = 0; i < 2; ++i) {
      const int nt = wv + 16 * i;       // 32 N-tiles over 16 waves
      const int n0 = nt * 16;
      v8f acc = {0.f,0.f,0.f,0.f,0.f,0.f,0.f,0.f};
#pragma unroll
      for (int ks = 0; ks < 2; ++ks) {
        int k0 = ks * 32;
        v16bf a = ldsA(s_q + qoff, KD, ml, k0, half);
        v16bf b = ldB_keysT(keys, k0, n0 + nl, half);
        acc = __builtin_amdgcn_wmma_f32_16x16x32_bf16(false, a, false, b, (short)0, acc, false, false);
      }
#pragma unroll
      for (int r = 0; r < 8; ++r) {
        int m = r + 8 * half;
        sc[m * 512 + n0 + nl] = acc[r];
      }
    }
  };

  // ---------------- Row scores + per-row top4 ----------------
  score_pass(row_keys, 0);
  __syncthreads();

  float rv[4]; int ri[4];
  if (tid < MTILE) top4_scan(sc + tid * 512, rv, ri);
  __syncthreads();

  // ---------------- Col scores + combine (product-key top-k) ----------------
  score_pass(col_keys, 64);
  __syncthreads();

  if (tid < MTILE) {
    float cv[4]; int ci[4];
    top4_scan(sc + tid * 512, cv, ci);
    float fv[4] = {-3.4e38f, -3.4e38f, -3.4e38f, -3.4e38f};
    int   fi[4] = {0, 0, 0, 0};
#pragma unroll
    for (int i = 0; i < 4; ++i)
#pragma unroll
      for (int j = 0; j < 4; ++j)
        ins4(fv, fi, rv[i] + cv[j], ri[i] * NKEYS + ci[j]);
    const int g = m0 + tid;
#pragma unroll
    for (int k = 0; k < 4; ++k) {
      out_idx[g * 4 + k] = (float)fi[k];
      out_sc [g * 4 + k] = fv[k];
      s_tki[tid * 4 + k] = fi[k];
    }
  }
  __syncthreads();

  // ---------------- Projection: queries @ Wqc + bqc (M=16, N=768, K=128) ----------------
#pragma unroll
  for (int i = 0; i < 3; ++i) {
    const int nt = wv + 16 * i;         // 48 N-tiles over 16 waves
    const int n0 = nt * 16;
    v8f acc = {0.f,0.f,0.f,0.f,0.f,0.f,0.f,0.f};
#pragma unroll
    for (int ks = 0; ks < 4; ++ks) {
      int k0 = ks * 32;
      v16bf a = ldsA(s_q, KD, ml, k0, half);
      v16bf b = ldB_rowmajor(Wqc, DIM, k0, n0 + nl, half);
      acc = __builtin_amdgcn_wmma_f32_16x16x32_bf16(false, a, false, b, (short)0, acc, false, false);
    }
#pragma unroll
    for (int r = 0; r < 8; ++r) {
      int m = r + 8 * half;
      out3[(size_t)(m0 + m) * DIM + n0 + nl] = acc[r] + bqc[n0 + nl];
    }
  }

  // ---------------- Gather selected concept rows (float4 copies) ----------------
  {
    const int sel = tid >> 3;           // 64 selections (16 rows x 4)
    const int sub = tid & 7;
    const int m = sel >> 2, k = sel & 3;
    const int cidx = s_tki[sel];
    const float4* src = (const float4*)(concepts + (size_t)cidx * DIM);
    float4* dst = (float4*)(out0 + ((size_t)(m0 + m) * 4 + k) * DIM);
    for (int d4 = sub; d4 < DIM / 4; d4 += 8) dst[d4] = src[d4];
  }
}

extern "C" void kernel_launch(void* const* d_in, const int* in_sizes, int n_in,
                              void* d_out, int out_size, void* d_ws, size_t ws_size,
                              hipStream_t stream) {
  const float* regs     = (const float*)d_in[0];
  const float* row_keys = (const float*)d_in[1];
  const float* col_keys = (const float*)d_in[2];
  const float* concepts = (const float*)d_in[3];
  const float* gamma    = (const float*)d_in[4];
  const float* beta     = (const float*)d_in[5];
  const float* Wq       = (const float*)d_in[6];
  const float* bq       = (const float*)d_in[7];
  const float* Wqc      = (const float*)d_in[8];
  const float* bqc      = (const float*)d_in[9];

  float* out = (float*)d_out;
  // Tuple layout: concepts (4*256*768) | topk_idx (1024) | topk_scores (1024) | query_projected (4*64*768)
  float* out0    = out;
  float* out_idx = out0 + (size_t)4 * 256 * 768;
  float* out_sc  = out_idx + 1024;
  float* out3    = out_sc + 1024;

  concept_bank_fused<<<16, NTHREADS, 0, stream>>>(
      regs, row_keys, col_keys, concepts, gamma, beta, Wq, bq, Wqc, bqc,
      out0, out_idx, out_sc, out3);
}
